// basic_block_mod_69054484185133
// MI455X (gfx1250) — matrix operand order repaired and checked by reading, not compiled, not run on hardware
//
#include <hip/hip_runtime.h>
#include <math.h>
#include <stdint.h>

#define NNODE 160
#define NBATCH 4
#define NN (NNODE*NNODE)        // 25600
#define M_EDGES (NBATCH*NN)     // 102400
#define DIN 128
#define OUTD 384
#define ROWS (NBATCH*NNODE)     // 640

typedef __attribute__((ext_vector_type(16))) _Float16 v16h;
typedef __attribute__((ext_vector_type(8)))  _Float16 h8;
typedef __attribute__((ext_vector_type(8)))  float    v8f;

__device__ __forceinline__ float lrelu_f(float x){ return x > 0.f ? x : 0.01f*x; }
__device__ __forceinline__ float sigm_f(float x){ return 1.f/(1.f+expf(-x)); }

// ---------------------------------------------------------------- utilities
__global__ void k_zero(float* p, int n){
  int i = blockIdx.x*blockDim.x+threadIdx.x;
  if (i < n) p[i] = 0.f;
}

__global__ void k_f16cvt(const float* __restrict__ src, _Float16* __restrict__ dst, int n){
  int i = blockIdx.x*blockDim.x+threadIdx.x;
  if (i < n) dst[i] = (_Float16)src[i];
}

// t[b,i,d] = sum_j A[b,i,j] * X[b,j,d]
__global__ void k_bmm(const float* __restrict__ A, const float* __restrict__ X,
                      float* __restrict__ T){
  int idx = blockIdx.x*blockDim.x+threadIdx.x;
  if (idx >= NBATCH*NNODE*DIN) return;
  int d = idx % DIN;
  int i = (idx / DIN) % NNODE;
  int b = idx / (DIN*NNODE);
  const float* arow = A + (size_t)b*NN + (size_t)i*NNODE;
  const float* xcol = X + (size_t)b*NNODE*DIN + d;
  float s = 0.f;
  for (int j = 0; j < NNODE; ++j) s += arow[j]*xcol[(size_t)j*DIN];
  T[idx] = s;
}

// out[r,o] = dot(in[r,:], W[o,:]) + b[o]; accumulate per-channel sum/sumsq
__global__ void k_fc_stats(const float* __restrict__ in, const float* __restrict__ W,
                           const float* __restrict__ bias, float* __restrict__ out,
                           float* __restrict__ stats, int rows, int din, int dout){
  int idx = blockIdx.x*blockDim.x+threadIdx.x;
  if (idx >= rows*dout) return;
  int o = idx % dout;
  int r = idx / dout;
  const float* xr = in + (size_t)r*din;
  const float* wr = W  + (size_t)o*din;
  float s = bias[o];
  for (int k = 0; k < din; ++k) s += xr[k]*wr[k];
  out[idx] = s;
  atomicAdd(&stats[2*o],   s);
  atomicAdd(&stats[2*o+1], s*s);
}

// per-channel BN affine fold: y = a*x + c with a = g*rsqrt(var+eps), c = be - mean*a
__global__ void k_affine(const float* __restrict__ stats, const float* __restrict__ g,
                         const float* __restrict__ be, float* __restrict__ aff,
                         int ch, float inv_count){
  int c = blockIdx.x*blockDim.x+threadIdx.x;
  if (c >= ch) return;
  float m = stats[2*c]*inv_count;
  float v = stats[2*c+1]*inv_count - m*m;
  float a = g[c]*rsqrtf(v + 1e-5f);
  aff[2*c]   = a;
  aff[2*c+1] = be[c] - m*a;
}

__global__ void k_bn_lrelu(float* __restrict__ h, const float* __restrict__ aff,
                           int total, int ch){
  int idx = blockIdx.x*blockDim.x+threadIdx.x;
  if (idx >= total) return;
  int c = idx % ch;
  h[idx] = lrelu_f(aff[2*c]*h[idx] + aff[2*c+1]);
}

// gates + xcat (f32 + f16 copy) + Wcat output
__global__ void k_xcat(const float* __restrict__ x, const float* __restrict__ x1,
                       const float* __restrict__ x2,
                       const float* __restrict__ W0, const float* __restrict__ b0,
                       const float* __restrict__ W1, const float* __restrict__ b1,
                       const float* __restrict__ W2, const float* __restrict__ b2,
                       float* __restrict__ Wcat, float* __restrict__ XC,
                       _Float16* __restrict__ XH){
  int r = blockIdx.x*blockDim.x+threadIdx.x;
  if (r >= ROWS) return;
  const float* xr  = x  + (size_t)r*DIN;
  const float* x1r = x1 + (size_t)r*DIN;
  const float* x2r = x2 + (size_t)r*DIN;
  float s0 = b0[0], s1 = b1[0], s2 = b2[0];
  for (int k = 0; k < DIN; ++k){ s0 += xr[k]*W0[k]; s1 += x1r[k]*W1[k]; s2 += x2r[k]*W2[k]; }
  float w0 = sigm_f(s0), w1 = sigm_f(s1), w2 = sigm_f(s2);
  Wcat[r*3+0] = w0; Wcat[r*3+1] = w1; Wcat[r*3+2] = w2;
  float*    xc = XC + (size_t)r*OUTD;
  _Float16* xh = XH + (size_t)r*OUTD;
  for (int k = 0; k < DIN; ++k){
    float v0 = w0*xr[k], v1 = w1*x1r[k], v2 = w2*x2r[k];
    xc[k] = v0; xc[DIN+k] = v1; xc[2*DIN+k] = v2;
    xh[k] = (_Float16)v0; xh[DIN+k] = (_Float16)v1; xh[2*DIN+k] = (_Float16)v2;
  }
}

// ------------------------------------------------- WMMA edge-MLP GEMM layers
// One block (128 thr = 4 waves) owns one 16-edge m-tile. A-tile (16 x Kin f16)
// staged in LDS once (MODE 0: |xcat_i - xcat_j| computed while staging; MODE 1:
// raw prev-layer Z copied with global_load_async_to_lds_b128, BN-affine+lrelu
// applied at fragment build). Waves sweep the Nout/16 column tiles; each
// fragment is two contiguous 16B chunks per lane per the ISA VGPR layouts.
template<int MODE>
__global__ void k_edge_gemm(const _Float16* __restrict__ XH,
                            const _Float16* __restrict__ Zin,
                            const float* __restrict__ affIn,
                            const _Float16* __restrict__ Wh,
                            const float* __restrict__ bias,
                            _Float16* __restrict__ Zout,
                            float* __restrict__ stats,
                            int Kin, int Nout){
  __shared__ __align__(16) _Float16 shA[16*OUTD];   // 12 KB max
  const int tid  = threadIdx.x;                     // 0..127
  const int wave = tid >> 5;
  const int lane = tid & 31;
  const int hs   = lane >> 4;
  const int l15  = lane & 15;
  const int m0   = blockIdx.x * 16;

  const int chunks = (16*Kin) >> 3;                 // 8-half (16B) chunks
  if (MODE == 0) {
    for (int c = tid; c < chunks; c += 128) {
      int row = c / (Kin >> 3);
      int kc  = (c - row*(Kin >> 3)) << 3;
      int m = m0 + row;
      int b = m / NN; int rem = m - b*NN;
      int ei = rem / NNODE; int ej = rem - ei*NNODE;
      h8 va = *(const h8*)(XH + ((size_t)(b*NNODE+ei))*OUTD + kc);
      h8 vb = *(const h8*)(XH + ((size_t)(b*NNODE+ej))*OUTD + kc);
      h8 d;
#pragma unroll
      for (int e = 0; e < 8; ++e){
        _Float16 t = va[e] - vb[e];
        d[e] = (t < (_Float16)0.f) ? (_Float16)(-t) : t;
      }
      *(h8*)(shA + row*Kin + kc) = d;
    }
  } else {
    // pure copy: CDNA5 async global->LDS path (ASYNCcnt-tracked)
    for (int c = tid; c < chunks; c += 128) {
      int row = c / (Kin >> 3);
      int kc  = (c - row*(Kin >> 3)) << 3;
      const _Float16* g = Zin + (size_t)(m0+row)*Kin + kc;
      unsigned loff = (unsigned)(uintptr_t)(shA + row*Kin + kc); // LDS byte offset = addr[31:0]
      asm volatile("global_load_async_to_lds_b128 %0, %1, off"
                   :: "v"(loff), "v"(g) : "memory");
    }
    asm volatile("s_wait_asynccnt 0x0" ::: "memory");
  }
  __syncthreads();

  const int ntiles = Nout >> 4;
  for (int nt = wave; nt < ntiles; nt += 4) {
    const int n0 = nt << 4;
    const int nB = n0 + l15;
    const _Float16* wr = Wh + (size_t)nB * Kin;
    v8f acc = {0.f,0.f,0.f,0.f,0.f,0.f,0.f,0.f};
    for (int k0 = 0; k0 < Kin; k0 += 32) {
      __builtin_prefetch(wr + k0 + 64, 0, 0);
      // A fragment: lane = row l15; chunk0 K=k0+8*hs.., chunk1 K=k0+16+8*hs..
      h8 a0 = *(const h8*)(shA + l15*Kin + k0 + hs*8);
      h8 a1 = *(const h8*)(shA + l15*Kin + k0 + 16 + hs*8);
      v16h af;
      if (MODE == 0) {
#pragma unroll
        for (int e = 0; e < 8; ++e){ af[e] = a0[e]; af[8+e] = a1[e]; }
      } else {
        const float* ap0 = affIn + 2*(k0 + hs*8);
        const float* ap1 = affIn + 2*(k0 + 16 + hs*8);
#pragma unroll
        for (int e = 0; e < 8; ++e){
          float y0 = ap0[2*e]*(float)a0[e] + ap0[2*e+1];
          float y1 = ap1[2*e]*(float)a1[e] + ap1[2*e+1];
          af[e]   = (_Float16)(y0 > 0.f ? y0 : 0.01f*y0);
          af[8+e] = (_Float16)(y1 > 0.f ? y1 : 0.01f*y1);
        }
      }
      // B fragment: lane = col nB; chunk0 K=k0+8*hs.., chunk1 K=k0+16+8*hs..
      h8 b0 = *(const h8*)(wr + k0 + hs*8);
      h8 b1 = *(const h8*)(wr + k0 + 16 + hs*8);
      v16h bf;
#pragma unroll
      for (int e = 0; e < 8; ++e){ bf[e] = b0[e]; bf[8+e] = b1[e]; }
      acc = __builtin_amdgcn_wmma_f32_16x16x32_f16(false, af, false, bf,
                                                   (short)0, acc, false, false);
    }
    // D layout: lane holds column nB; VGPR v holds row m0 + 8*hs + v
    float bn = bias[nB];
    float s = 0.f, ss = 0.f;
#pragma unroll
    for (int v = 0; v < 8; ++v) {
      float val = acc[v] + bn;
      int m = m0 + 8*hs + v;
      Zout[(size_t)m * Nout + nB] = (_Float16)val;
      s += val; ss += val*val;
    }
    atomicAdd(&stats[2*nB],   s);
    atomicAdd(&stats[2*nB+1], ss);
  }
}

// final 96->1 layer + sigmoid + diagonal mask
__global__ void k_edge_last(const _Float16* __restrict__ Zin, const float* __restrict__ aff,
                            const float* __restrict__ Wl, const float* __restrict__ bl,
                            float* __restrict__ Aout){
  int e = blockIdx.x*blockDim.x+threadIdx.x;
  if (e >= M_EDGES) return;
  const _Float16* zrow = Zin + (size_t)e * 96;
  float s = bl[0];
  for (int k = 0; k < 96; ++k){
    float y = aff[2*k]*(float)zrow[k] + aff[2*k+1];
    s += lrelu_f(y) * Wl[k];
  }
  int rem = e % NN; int i = rem / NNODE; int j = rem % NNODE;
  Aout[e] = (i == j) ? 0.f : sigm_f(s);
}

__global__ void k_dinv(const float* __restrict__ Aout, float* __restrict__ dinv){
  int r = blockIdx.x*blockDim.x+threadIdx.x;
  if (r >= ROWS) return;
  int b = r / NNODE; int i = r % NNODE;
  const float* row = Aout + (size_t)b*NN + (size_t)i*NNODE;
  float s = 1.0f;                       // +I on the diagonal (A[ii]==0)
  for (int j = 0; j < NNODE; ++j) s += row[j];
  dinv[r] = rsqrtf(s);
}

__global__ void k_anorm(const float* __restrict__ Aout, const float* __restrict__ dinv,
                        float* __restrict__ Anout){
  int e = blockIdx.x*blockDim.x+threadIdx.x;
  if (e >= M_EDGES) return;
  int b = e / NN; int rem = e % NN; int i = rem / NNODE; int j = rem % NNODE;
  float a = Aout[e] + ((i == j) ? 1.f : 0.f);
  Anout[e] = dinv[b*NNODE+i] * a * dinv[b*NNODE+j];
}

__global__ void k_xout(const float* __restrict__ XC, const float* __restrict__ origpre,
                       const float* __restrict__ aff, float* __restrict__ xout){
  int idx = blockIdx.x*blockDim.x+threadIdx.x;
  if (idx >= ROWS*OUTD) return;
  int c = idx % OUTD;
  xout[idx] = lrelu_f(XC[idx] + aff[2*c]*origpre[idx] + aff[2*c+1]);
}

// ---------------------------------------------------------------- launcher
extern "C" void kernel_launch(void* const* d_in, const int* in_sizes, int n_in,
                              void* d_out, int out_size, void* d_ws, size_t ws_size,
                              hipStream_t stream)
{
  (void)in_sizes; (void)n_in; (void)out_size; (void)ws_size;
  // top-level dict order: x, A_norm, params; params leaves in pytree (sorted-key) order
  const float* x      = (const float*)d_in[0];
  const float* A_norm = (const float*)d_in[1];
  const float* adjLW  = (const float*)d_in[2];   // adj.last.W  (1,96)
  const float* adjLb  = (const float*)d_in[3];   // adj.last.b
  const float* aW[4]  = {(const float*)d_in[4],  (const float*)d_in[8],
                         (const float*)d_in[12], (const float*)d_in[16]};
  const float* ab[4]  = {(const float*)d_in[5],  (const float*)d_in[9],
                         (const float*)d_in[13], (const float*)d_in[17]};
  const float* abe[4] = {(const float*)d_in[6],  (const float*)d_in[10],
                         (const float*)d_in[14], (const float*)d_in[18]};
  const float* ag[4]  = {(const float*)d_in[7],  (const float*)d_in[11],
                         (const float*)d_in[15], (const float*)d_in[19]};
  const float* g1W  = (const float*)d_in[20]; const float* g1b  = (const float*)d_in[21];
  const float* g1be = (const float*)d_in[22]; const float* g1g  = (const float*)d_in[23];
  const float* g2aW = (const float*)d_in[24]; const float* g2ab = (const float*)d_in[25];
  const float* g2abe= (const float*)d_in[26]; const float* g2ag = (const float*)d_in[27];
  const float* g2bW = (const float*)d_in[28]; const float* g2bb = (const float*)d_in[29];
  const float* g2bbe= (const float*)d_in[30]; const float* g2bg = (const float*)d_in[31];
  const float* trW  = (const float*)d_in[32]; const float* trb  = (const float*)d_in[33];
  const float* trbe = (const float*)d_in[34]; const float* trg  = (const float*)d_in[35];
  const float* w0W  = (const float*)d_in[36]; const float* w0b  = (const float*)d_in[37];
  const float* w1W  = (const float*)d_in[38]; const float* w1b  = (const float*)d_in[39];
  const float* w2W  = (const float*)d_in[40]; const float* w2b  = (const float*)d_in[41];

  float* out = (float*)d_out;
  float* oA  = out;               // [4,160,160]
  float* oAn = out + M_EDGES;     // [4,160,160]
  float* oW  = out + 2*M_EDGES;   // [4,160,3]
  float* oX  = out + 2*M_EDGES + ROWS*3;  // [4,160,384]

  float* ws = (float*)d_ws;
  size_t o = 0;
  float* T    = ws + o; o += 81920;
  float* H1   = ws + o; o += 81920;     // -> x1 in place
  float* H2A  = ws + o; o += 81920;     // -> x2a in place
  float* H2B  = ws + o; o += 81920;     // -> x2 in place
  float* ORIG = ws + o; o += 245760;    // trans pre-BN
  float* XC   = ws + o; o += 245760;    // xcat f32
  _Float16* XHh = (_Float16*)(ws + o); o += 122880;  // xcat f16 (245760 halves)
  _Float16* WH0 = (_Float16*)(ws + o); o += 36864;   // 73728 halves
  _Float16* WH1 = (_Float16*)(ws + o); o += 18432;   // 36864 halves
  _Float16* WH2 = (_Float16*)(ws + o); o += 9216;    // 18432 halves
  _Float16* WH3 = (_Float16*)(ws + o); o += 4608;    //  9216 halves
  float* STAT = ws + o; o += 4096;
  float* AFF  = ws + o; o += 4096;
  float* DINV = ws + o; o += 1024;
  _Float16* ZP = (_Float16*)(ws + o); o += 9830400;  // 102400*192 halves
  _Float16* ZQ = (_Float16*)(ws + o); o += 9830400;

  float* st_g1  = STAT;        float* af_g1  = AFF;
  float* st_g2a = STAT + 256;  float* af_g2a = AFF + 256;
  float* st_g2b = STAT + 512;  float* af_g2b = AFF + 512;
  float* st_tr  = STAT + 768;  float* af_tr  = AFF + 768;
  float* st_a0  = STAT + 1536; float* af_a0  = AFF + 1536;
  float* st_a1  = STAT + 1920; float* af_a1  = AFF + 1920;
  float* st_a2  = STAT + 2304; float* af_a2  = AFF + 2304;
  float* st_a3  = STAT + 2496; float* af_a3  = AFF + 2496;

  const int TB = 256;
  #define GRD(n) dim3(((n)+TB-1)/TB), dim3(TB)

  k_zero<<<GRD(4096), 0, stream>>>(STAT, 4096);
  k_f16cvt<<<GRD(73728), 0, stream>>>(aW[0], WH0, 73728);
  k_f16cvt<<<GRD(36864), 0, stream>>>(aW[1], WH1, 36864);
  k_f16cvt<<<GRD(18432), 0, stream>>>(aW[2], WH2, 18432);
  k_f16cvt<<<GRD(9216),  0, stream>>>(aW[3], WH3, 9216);

  // branch 1: one gconv
  k_bmm<<<GRD(81920), 0, stream>>>(A_norm, x, T);
  k_fc_stats<<<GRD(640*128), 0, stream>>>(T, g1W, g1b, H1, st_g1, ROWS, DIN, DIN);
  k_affine<<<GRD(128), 0, stream>>>(st_g1, g1g, g1be, af_g1, 128, 1.f/ROWS);
  k_bn_lrelu<<<GRD(81920), 0, stream>>>(H1, af_g1, 81920, 128);

  // branch 2: two gconvs
  k_bmm<<<GRD(81920), 0, stream>>>(A_norm, x, T);
  k_fc_stats<<<GRD(640*128), 0, stream>>>(T, g2aW, g2ab, H2A, st_g2a, ROWS, DIN, DIN);
  k_affine<<<GRD(128), 0, stream>>>(st_g2a, g2ag, g2abe, af_g2a, 128, 1.f/ROWS);
  k_bn_lrelu<<<GRD(81920), 0, stream>>>(H2A, af_g2a, 81920, 128);
  k_bmm<<<GRD(81920), 0, stream>>>(A_norm, H2A, T);
  k_fc_stats<<<GRD(640*128), 0, stream>>>(T, g2bW, g2bb, H2B, st_g2b, ROWS, DIN, DIN);
  k_affine<<<GRD(128), 0, stream>>>(st_g2b, g2bg, g2bbe, af_g2b, 128, 1.f/ROWS);
  k_bn_lrelu<<<GRD(81920), 0, stream>>>(H2B, af_g2b, 81920, 128);

  // trans (pre-BN kept raw; affine folded at the very end)
  k_fc_stats<<<GRD(640*384), 0, stream>>>(x, trW, trb, ORIG, st_tr, ROWS, DIN, OUTD);
  k_affine<<<GRD(384), 0, stream>>>(st_tr, trg, trbe, af_tr, 384, 1.f/ROWS);

  // gates + xcat (+Wcat output)
  k_xcat<<<GRD(ROWS), 0, stream>>>(x, H1, H2B, w0W, w0b, w1W, w1b, w2W, w2b,
                                   oW, XC, XHh);

  // edge MLP: 4 WMMA layers with two-pass BN; 128-thread blocks, 1 m-tile/block
  dim3 gblk(128);
  dim3 ggrid(M_EDGES/16);
  k_edge_gemm<0><<<ggrid, gblk, 0, stream>>>(
      XHh, nullptr, nullptr, WH0, ab[0], ZP, st_a0, 384, 192);
  k_affine<<<GRD(192), 0, stream>>>(st_a0, ag[0], abe[0], af_a0, 192, 1.f/M_EDGES);
  k_edge_gemm<1><<<ggrid, gblk, 0, stream>>>(
      nullptr, ZP, af_a0, WH1, ab[1], ZQ, st_a1, 192, 192);
  k_affine<<<GRD(192), 0, stream>>>(st_a1, ag[1], abe[1], af_a1, 192, 1.f/M_EDGES);
  k_edge_gemm<1><<<ggrid, gblk, 0, stream>>>(
      nullptr, ZQ, af_a1, WH2, ab[2], ZP, st_a2, 192, 96);
  k_affine<<<GRD(96), 0, stream>>>(st_a2, ag[2], abe[2], af_a2, 96, 1.f/M_EDGES);
  k_edge_gemm<1><<<ggrid, gblk, 0, stream>>>(
      nullptr, ZP, af_a2, WH3, ab[3], ZQ, st_a3, 96, 96);
  k_affine<<<GRD(96), 0, stream>>>(st_a3, ag[3], abe[3], af_a3, 96, 1.f/M_EDGES);

  k_edge_last<<<GRD(M_EDGES), 0, stream>>>(ZQ, af_a3, adjLW, adjLb, oA);
  k_dinv<<<GRD(ROWS), 0, stream>>>(oA, DINV);
  k_anorm<<<GRD(M_EDGES), 0, stream>>>(oA, DINV, oAn);
  k_xout<<<GRD(ROWS*OUTD), 0, stream>>>(XC, ORIG, af_tr, oX);
  #undef GRD
}
